// BidirectionalMambaBlock_58342835749243
// MI455X (gfx1250) — matrix operand order repaired and checked by reading, not compiled, not run on hardware
//
#include <hip/hip_runtime.h>
#include <hip/hip_bf16.h>

// ---------------------------------------------------------------------------
// BidirectionalMambaBlock for MI455X (gfx1250, wave32, WMMA)
// All GEMMs via v_wmma_f32_16x16x32_f16 (f32 accumulate).
// ---------------------------------------------------------------------------

typedef __attribute__((ext_vector_type(16))) _Float16 v16h;
typedef __attribute__((ext_vector_type(8)))  _Float16 v8h;
typedef __attribute__((ext_vector_type(8)))  float    v8f;

#define BSZ  4
#define DIM  96
#define LL   4096      // H*W
#define MTOT 16384     // BSZ*LL
#define MD   32768     // 2 dirs * MTOT
#define DI   192
#define NST  16
#define DTR  6
#define NDBC 48        // 38 padded to 48

// ---------------- workspace layout (floats then halves) --------------------
#define F_UPRE  ((size_t)0)                    // [2][MTOT][DI]  (aliased by ys after conv)
#define F_U     (F_UPRE + (size_t)MD*DI)
#define F_Z     (F_U    + (size_t)MD*DI)
#define F_DBC   (F_Z    + (size_t)MD*DI)       // [2][MTOT][48]
#define F_DELTA (F_DBC  + (size_t)MD*NDBC)
#define F_ODIR  (F_DELTA+ (size_t)MD*DI)       // [2][MTOT][96]
#define F_YF    (F_ODIR + (size_t)MD*DIM)      // [MTOT][96]
#define F_STAT  (F_YF   + (size_t)MTOT*DIM)    // mean[96], var[96]
#define F_ANEG  (F_STAT + 192)                 // [2][DI][NST]
#define F_END   (F_ANEG + 2*DI*NST)
#define H_INW   ((size_t)0)                    // [2][96][384]
#define H_XPW   (H_INW + 2*96*384)             // [2][192][48]
#define H_DTW   (H_XPW + 2*192*48)             // [2][32][192]
#define H_OUTW  (H_DTW + 2*32*192)             // [2][192][96]
#define H_FUSE  (H_OUTW + 2*192*96)            // [192][96]

// ---------------- WMMA helpers (ISA 7.12.2 layouts) ------------------------
union U16h { v16h v; v8h p[2]; };

__device__ __forceinline__ v16h load_a_frag(const _Float16* As, int lda, int lane, int kbase) {
    int m = lane & 15, hh = lane >> 4;
    U16h u;
    u.p[0] = *(const v8h*)(As + m * lda + kbase + hh * 8);        // K = hh*8 + 0..7
    u.p[1] = *(const v8h*)(As + m * lda + kbase + 16 + hh * 8);   // K = 16 + hh*8 + 0..7
    return u.v;
}
__device__ __forceinline__ v16h load_b_frag(const _Float16* Bs, int ldb, int lane, int kbase) {
    int n = lane & 15, hh = lane >> 4;
    U16h u;
    u.p[0] = *(const v8h*)(Bs + n * ldb + kbase + hh * 8);        // K = hh*8 + 0..7
    u.p[1] = *(const v8h*)(Bs + n * ldb + kbase + 16 + hh * 8);   // K = 16 + hh*8 + 0..7
    return u.v;
}
__device__ __forceinline__ v8f wmma32(v16h a, v16h b, v8f c) {
    return __builtin_amdgcn_wmma_f32_16x16x32_f16(false, a, false, b, (short)0, c, false, false);
}
__device__ __forceinline__ float sigf(float x) { return 1.f / (1.f + __expf(-x)); }

// ---------------- 0: weight convert / prep ---------------------------------
__global__ void k_prep(const float* fInW, const float* bInW,
                       const float* fXpW, const float* bXpW,
                       const float* fDtW, const float* bDtW,
                       const float* fAlog, const float* bAlog,
                       const float* fOutW, const float* bOutW,
                       const float* fuseW,
                       _Float16* hb, float* Aneg) {
    int tid = blockIdx.x * blockDim.x + threadIdx.x;
    int stride = gridDim.x * blockDim.x;
    _Float16* inW  = hb + H_INW;
    _Float16* xpW  = hb + H_XPW;
    _Float16* dtW  = hb + H_DTW;
    _Float16* outW = hb + H_OUTW;
    _Float16* fusW = hb + H_FUSE;
    for (int i = tid; i < 96 * 384; i += stride) {
        inW[i] = (_Float16)fInW[i];
        inW[96 * 384 + i] = (_Float16)bInW[i];
    }
    for (int i = tid; i < 192 * 48; i += stride) {
        int k = i / 48, n = i % 48;
        xpW[i]            = (n < 38) ? (_Float16)fXpW[k * 38 + n] : (_Float16)0.f;
        xpW[192 * 48 + i] = (n < 38) ? (_Float16)bXpW[k * 38 + n] : (_Float16)0.f;
    }
    for (int i = tid; i < 32 * 192; i += stride) {
        int k = i / 192, n = i % 192;
        dtW[i]             = (k < DTR) ? (_Float16)fDtW[k * 192 + n] : (_Float16)0.f;
        dtW[32 * 192 + i]  = (k < DTR) ? (_Float16)bDtW[k * 192 + n] : (_Float16)0.f;
    }
    for (int i = tid; i < 192 * 96; i += stride) {
        outW[i]            = (_Float16)fOutW[i];
        outW[192 * 96 + i] = (_Float16)bOutW[i];
        int k = i / 96, n = i % 96;
        fusW[i] = (_Float16)fuseW[n * 192 + k];    // transpose [o][c] -> [c][o]
    }
    for (int i = tid; i < DI * NST; i += stride) {
        Aneg[i]            = -__expf(fAlog[i]);
        Aneg[DI * NST + i] = -__expf(bAlog[i]);
    }
}

// ---------------- 1: in-projection GEMM  (M=16384, K=96, N=384) ------------
__global__ __launch_bounds__(32) void k_inproj(const float* x, const _Float16* hb,
                                               float* u_pre, float* z) {
    __shared__ alignas(16) _Float16 As[16][96];
    __shared__ alignas(16) _Float16 Bs[16][96];   // [n][k]
    int lane = threadIdx.x;
    int mt = blockIdx.x, nt = blockIdx.y, dir = blockIdx.z;
    int b = (mt * 16) >> 12;
    const float* xb = x + (size_t)b * DIM * LL;
    for (int i = lane; i < 16 * 96; i += 32) {
        int r = i / 96, c = i % 96;
        int l = mt * 16 + r - b * LL;
        int xl = dir ? (LL - 1 - l) : l;
        As[r][c] = (_Float16)xb[(size_t)c * LL + xl];
    }
    const _Float16* w = hb + H_INW + (size_t)dir * 96 * 384 + nt * 16;
    for (int i = lane; i < 16 * 96; i += 32) {
        int n = i & 15, k = i >> 4;
        Bs[n][k] = w[k * 384 + n];
    }
    __syncthreads();
    v8f acc = {};
#pragma unroll
    for (int kt = 0; kt < 3; ++kt)
        acc = wmma32(load_a_frag(&As[0][0], 96, lane, kt * 32),
                     load_b_frag(&Bs[0][0], 96, lane, kt * 32), acc);
    int n = lane & 15, hh = lane >> 4;
    int j = nt * 16 + n;
    // branch-free: select destination base once (j is invariant over r),
    // then 8 unconditional stores -> no exec-mask ladder.
    size_t m0 = (size_t)dir * MTOT + mt * 16 + hh * 8;
    float* dst = (j < DI) ? (u_pre + m0 * DI + j) : (z + m0 * DI + (j - DI));
#pragma unroll
    for (int r = 0; r < 8; ++r)
        dst[(size_t)r * DI] = acc[r];
}

// ---------------- 2: depthwise causal conv (K=4) + SiLU --------------------
__global__ void k_conv(const float* u_pre, const float* cwF, const float* cbF,
                       const float* cwB, const float* cbB, float* u) {
    size_t idx = (size_t)blockIdx.x * blockDim.x + threadIdx.x;
    if (idx >= (size_t)MD * DI) return;
    int d = idx % DI;
    size_t mi = idx / DI;
    int dir = (int)(mi / MTOT);
    int l = (int)(mi & (LL - 1));
    const float* cw = dir ? cwB : cwF;
    const float* cb = dir ? cbB : cbF;
    float s = 0.f;
#pragma unroll
    for (int k = 0; k < 4; ++k) {
        int ll = l - 3 + k;
        if (ll >= 0) s += u_pre[(mi - 3 + k) * DI + d] * cw[d * 4 + k];
    }
    s += cb[d];
    u[idx] = s * sigf(s);   // SiLU
}

// ---------------- 3: x-projection GEMM (M=16384, K=192, N=48pad) -----------
__global__ __launch_bounds__(32) void k_xproj(const float* u, const _Float16* hb, float* dbc) {
    __shared__ alignas(16) _Float16 As[16][192];
    __shared__ alignas(16) _Float16 Bs[16][192];
    int lane = threadIdx.x;
    int mt = blockIdx.x, nt = blockIdx.y, dir = blockIdx.z;
    size_t m0 = (size_t)dir * MTOT + mt * 16;
    for (int i = lane; i < 16 * 192; i += 32) {
        int r = i / 192, c = i % 192;
        As[r][c] = (_Float16)u[(m0 + r) * DI + c];
    }
    const _Float16* w = hb + H_XPW + (size_t)dir * 192 * 48 + nt * 16;
    for (int i = lane; i < 16 * 192; i += 32) {
        int n = i & 15, k = i >> 4;
        Bs[n][k] = w[k * 48 + n];
    }
    __syncthreads();
    v8f acc = {};
#pragma unroll
    for (int kt = 0; kt < 6; ++kt)
        acc = wmma32(load_a_frag(&As[0][0], 192, lane, kt * 32),
                     load_b_frag(&Bs[0][0], 192, lane, kt * 32), acc);
    int n = lane & 15, hh = lane >> 4;
    int j = nt * 16 + n;
#pragma unroll
    for (int r = 0; r < 8; ++r)
        dbc[(m0 + r + hh * 8) * NDBC + j] = acc[r];
}

// ---------------- 4: delta GEMM (K=6 pad 32, N=192) + softplus -------------
__global__ __launch_bounds__(32) void k_delta(const float* dbc, const _Float16* hb,
                                              const float* dtbF, const float* dtbB, float* delta) {
    __shared__ alignas(16) _Float16 As[16][32];
    __shared__ alignas(16) _Float16 Bs[16][32];
    int lane = threadIdx.x;
    int mt = blockIdx.x, nt = blockIdx.y, dir = blockIdx.z;
    size_t m0 = (size_t)dir * MTOT + mt * 16;
    for (int i = lane; i < 16 * 32; i += 32) {
        int r = i / 32, c = i % 32;
        As[r][c] = (c < DTR) ? (_Float16)dbc[(m0 + r) * NDBC + c] : (_Float16)0.f;
    }
    const _Float16* w = hb + H_DTW + (size_t)dir * 32 * 192 + nt * 16;
    for (int i = lane; i < 16 * 32; i += 32) {
        int n = i & 15, k = i >> 4;
        Bs[n][k] = w[k * 192 + n];
    }
    __syncthreads();
    v8f acc = {};
    acc = wmma32(load_a_frag(&As[0][0], 32, lane, 0),
                 load_b_frag(&Bs[0][0], 32, lane, 0), acc);
    const float* dtb = dir ? dtbB : dtbF;
    int n = lane & 15, hh = lane >> 4;
    int j = nt * 16 + n;
#pragma unroll
    for (int r = 0; r < 8; ++r) {
        float xv = acc[r] + dtb[j];
        float sp = (xv > 20.f) ? xv : __logf(1.f + __expf(xv));   // softplus
        delta[(m0 + r + hh * 8) * DI + j] = sp;
    }
}

// ---------------- 5: selective scan (sequential over L) --------------------
// one 16-lane half-wave per (dir, b, channel d); lane n = state index
__global__ void k_scan(const float* delta, const float* u, const float* dbc,
                       const float* Aneg, const float* fD, const float* bD, float* ys) {
    int t = blockIdx.x * blockDim.x + threadIdx.x;
    int w = t >> 5, lane = t & 31;
    int n = lane & 15, hh = lane >> 4;
    int dir = w / (BSZ * DI / 2);
    int rem = w % (BSZ * DI / 2);
    int b = rem / (DI / 2);
    int d = (rem % (DI / 2)) * 2 + hh;
    float a_dn = Aneg[dir * DI * NST + d * NST + n];
    float Dd = (dir ? bD : fD)[d];
    size_t base = (size_t)dir * MTOT + (size_t)b * LL;
    float h = 0.f;
    for (int l = 0; l < LL; ++l) {
        size_t m = base + l;
        // prefetch a few rows ahead (global_prefetch_b8) to hide latency
        // behind the v_exp-dominated recurrence
        if (l + 8 < LL) {
            __builtin_prefetch(&delta[(m + 8) * DI + d], 0, 0);
            __builtin_prefetch(&dbc[(m + 8) * NDBC + n], 0, 0);
        }
        float dv = delta[m * DI + d];
        float uv = u[m * DI + d];
        float Bn = dbc[m * NDBC + DTR + n];
        float Cn = dbc[m * NDBC + DTR + NST + n];
        h = __expf(dv * a_dn) * h + dv * Bn * uv;          // h = dA*h + dBu
        float p = h * Cn;
        p += __shfl_xor(p, 1, 32);                          // reduce 16 states
        p += __shfl_xor(p, 2, 32);                          // (xor<16 stays in half)
        p += __shfl_xor(p, 4, 32);
        p += __shfl_xor(p, 8, 32);
        if (n == 0) ys[m * DI + d] = p + uv * Dd;
    }
}

// ---------------- 6: gate + out-projection GEMM (K=192, N=96) --------------
__global__ __launch_bounds__(32) void k_outproj(const float* ys, const float* z,
                                                const _Float16* hb, float* odir) {
    __shared__ alignas(16) _Float16 As[16][192];
    __shared__ alignas(16) _Float16 Bs[16][192];
    int lane = threadIdx.x;
    int mt = blockIdx.x, nt = blockIdx.y, dir = blockIdx.z;
    size_t m0 = (size_t)dir * MTOT + mt * 16;
    for (int i = lane; i < 16 * 192; i += 32) {
        int r = i / 192, c = i % 192;
        float zv = z[(m0 + r) * DI + c];
        As[r][c] = (_Float16)(ys[(m0 + r) * DI + c] * zv * sigf(zv));  // y * silu(z)
    }
    const _Float16* w = hb + H_OUTW + (size_t)dir * 192 * 96 + nt * 16;
    for (int i = lane; i < 16 * 192; i += 32) {
        int n = i & 15, k = i >> 4;
        Bs[n][k] = w[k * 96 + n];
    }
    __syncthreads();
    v8f acc = {};
#pragma unroll
    for (int kt = 0; kt < 6; ++kt)
        acc = wmma32(load_a_frag(&As[0][0], 192, lane, kt * 32),
                     load_b_frag(&Bs[0][0], 192, lane, kt * 32), acc);
    int n = lane & 15, hh = lane >> 4;
    int j = nt * 16 + n;
#pragma unroll
    for (int r = 0; r < 8; ++r)
        odir[(m0 + r + hh * 8) * DIM + j] = acc[r];
}

// ---------------- 7: fuse GEMM (concat fwd + flipped bwd, K=192, N=96) -----
__global__ __launch_bounds__(32) void k_fuse(const float* odir, const _Float16* hb,
                                             const float* fuse_b, float* yf) {
    __shared__ alignas(16) _Float16 As[16][192];
    __shared__ alignas(16) _Float16 Bs[16][192];
    int lane = threadIdx.x;
    int mt = blockIdx.x, nt = blockIdx.y;
    for (int i = lane; i < 16 * 192; i += 32) {
        int r = i / 192, c = i % 192;
        int m = mt * 16 + r;
        int b = m >> 12, l = m & (LL - 1);
        float v;
        if (c < DIM) v = odir[(size_t)m * DIM + c];
        else         v = odir[((size_t)MTOT + b * LL + (LL - 1 - l)) * DIM + (c - DIM)];
        As[r][c] = (_Float16)v;
    }
    const _Float16* w = hb + H_FUSE + nt * 16;
    for (int i = lane; i < 16 * 192; i += 32) {
        int n = i & 15, k = i >> 4;
        Bs[n][k] = w[k * 96 + n];
    }
    __syncthreads();
    v8f acc = {};
#pragma unroll
    for (int kt = 0; kt < 6; ++kt)
        acc = wmma32(load_a_frag(&As[0][0], 192, lane, kt * 32),
                     load_b_frag(&Bs[0][0], 192, lane, kt * 32), acc);
    int n = lane & 15, hh = lane >> 4;
    int j = nt * 16 + n;
#pragma unroll
    for (int r = 0; r < 8; ++r)
        yf[(size_t)(mt * 16 + r + hh * 8) * DIM + j] = acc[r] + fuse_b[j];
}

// ---------------- 8: batchnorm statistics ----------------------------------
__global__ void k_stats(const float* yf, float* stats) {
    int o = blockIdx.x;
    __shared__ float ssum[256], ssq[256];
    float s = 0.f, q = 0.f;
    for (int m = threadIdx.x; m < MTOT; m += 256) {
        float v = yf[(size_t)m * DIM + o];
        s += v; q += v * v;
    }
    ssum[threadIdx.x] = s; ssq[threadIdx.x] = q;
    __syncthreads();
    for (int st = 128; st > 0; st >>= 1) {
        if (threadIdx.x < st) {
            ssum[threadIdx.x] += ssum[threadIdx.x + st];
            ssq[threadIdx.x]  += ssq[threadIdx.x + st];
        }
        __syncthreads();
    }
    if (threadIdx.x == 0) {
        float mu = ssum[0] * (1.f / MTOT);
        stats[o] = mu;
        stats[DIM + o] = ssq[0] * (1.f / MTOT) - mu * mu;
    }
}

// ---------------- 9: normalize + relu + NCHW layout ------------------------
__global__ void k_norm(const float* yf, const float* stats, const float* g,
                       const float* bb, float* out) {
    size_t idx = (size_t)blockIdx.x * blockDim.x + threadIdx.x;
    if (idx >= (size_t)MTOT * DIM) return;
    int o = idx % DIM;
    int m = (int)(idx / DIM);
    int b = m >> 12, l = m & (LL - 1);
    float v = (yf[idx] - stats[o]) * rsqrtf(stats[DIM + o] + 1e-5f) * g[o] + bb[o];
    out[((size_t)b * DIM + o) * LL + l] = v > 0.f ? v : 0.f;
}

// ---------------------------------------------------------------------------
extern "C" void kernel_launch(void* const* d_in, const int* in_sizes, int n_in,
                              void* d_out, int out_size, void* d_ws, size_t ws_size,
                              hipStream_t stream) {
    (void)in_sizes; (void)n_in; (void)out_size; (void)ws_size;
    const float* x      = (const float*)d_in[0];
    const float* fInW   = (const float*)d_in[1];
    const float* fConvW = (const float*)d_in[2];
    const float* fConvB = (const float*)d_in[3];
    const float* fXpW   = (const float*)d_in[4];
    const float* fDtW   = (const float*)d_in[5];
    const float* fDtB   = (const float*)d_in[6];
    const float* fAlog  = (const float*)d_in[7];
    const float* fDp    = (const float*)d_in[8];
    const float* fOutW  = (const float*)d_in[9];
    const float* bInW   = (const float*)d_in[10];
    const float* bConvW = (const float*)d_in[11];
    const float* bConvB = (const float*)d_in[12];
    const float* bXpW   = (const float*)d_in[13];
    const float* bDtW   = (const float*)d_in[14];
    const float* bDtB   = (const float*)d_in[15];
    const float* bAlog  = (const float*)d_in[16];
    const float* bDp    = (const float*)d_in[17];
    const float* bOutW  = (const float*)d_in[18];
    const float* fuseW  = (const float*)d_in[19];
    const float* fuseB  = (const float*)d_in[20];
    const float* bnG    = (const float*)d_in[21];
    const float* bnB    = (const float*)d_in[22];
    float* out = (float*)d_out;

    float*     fb = (float*)d_ws;
    _Float16*  hb = (_Float16*)(fb + F_END);

    float* u_pre = fb + F_UPRE;
    float* u     = fb + F_U;
    float* z     = fb + F_Z;
    float* dbc   = fb + F_DBC;
    float* delta = fb + F_DELTA;
    float* ys    = fb + F_UPRE;    // alias: u_pre dead after conv
    float* odir  = fb + F_ODIR;
    float* yf    = fb + F_YF;
    float* stats = fb + F_STAT;
    float* Aneg  = fb + F_ANEG;

    k_prep<<<64, 256, 0, stream>>>(fInW, bInW, fXpW, bXpW, fDtW, bDtW,
                                   fAlog, bAlog, fOutW, bOutW, fuseW, hb, Aneg);
    k_inproj<<<dim3(MTOT / 16, 24, 2), 32, 0, stream>>>(x, hb, u_pre, z);
    k_conv<<<(MD * DI) / 256, 256, 0, stream>>>(u_pre, fConvW, fConvB, bConvW, bConvB, u);
    k_xproj<<<dim3(MTOT / 16, 3, 2), 32, 0, stream>>>(u, hb, dbc);
    k_delta<<<dim3(MTOT / 16, 12, 2), 32, 0, stream>>>(dbc, hb, fDtB, bDtB, delta);
    k_scan<<<(2 * BSZ * (DI / 2) * 32) / 256, 256, 0, stream>>>(delta, u, dbc, Aneg, fDp, bDp, ys);
    k_outproj<<<dim3(MTOT / 16, 6, 2), 32, 0, stream>>>(ys, z, hb, odir);
    k_fuse<<<dim3(MTOT / 16, 6), 32, 0, stream>>>(odir, hb, fuseB, yf);
    k_stats<<<DIM, 256, 0, stream>>>(yf, stats);
    k_norm<<<(MTOT * DIM) / 256, 256, 0, stream>>>(yf, stats, bnG, bnB, out);
}
